// RNNWithTransformer_54425825575653
// MI455X (gfx1250) — matrix operand order repaired and checked by reading, not compiled, not run on hardware
//
#include <hip/hip_runtime.h>
#include <hip/hip_bf16.h>
#include <math.h>

#define GRAPH_DIM   50257
#define EMBED_DIM   128
#define STATE_COUNT 5
#define N_STEP      512
#define BATCH       8
#define NROWS       (BATCH * N_STEP)   /* 4096 */

typedef __attribute__((ext_vector_type(16))) _Float16 v16h;
typedef __attribute__((ext_vector_type(8)))  float    v8f;

// TDM descriptor operand types (amdgpu-toolchain clang-23 6-arg builtin form:
//   (uint32x4 g0, int32x8 g1, int32x4 g2, int32x4 g3, int32x8, i32 cpol) )
typedef __attribute__((ext_vector_type(4))) unsigned int uint32x4;
typedef __attribute__((ext_vector_type(8))) int          int32x8;
typedef __attribute__((ext_vector_type(4))) int          int32x4;

// ---------------------------------------------------------------------------
// TDM: DMA a 2-D row-major f32 tile (rows x cols, stride == cols) from global
// memory into LDS at byte offset lds_off. D# layout per CDNA5 ISA ch.8:
//   group0: count=1 | lds_addr | global_addr[56:0] | type=2
//   group1: data_size=4B | tensor_dim0/1 | tile_dim0/1 | dim0_stride
//   groups 2/3: zero (2-D tensor)
// Must be executed by ONE wave (TDM ignores EXEC; it is a per-wave DMA).
// ---------------------------------------------------------------------------
__device__ __forceinline__ void tdm_load_2d_f32(const float* gptr,
                                                unsigned int lds_off,
                                                int rows, int cols) {
  unsigned long long ga = (unsigned long long)(uintptr_t)gptr;
  uint32x4 g0;
  g0[0] = 1u;                                            // count=1, user D#
  g0[1] = lds_off;                                       // lds_addr (bytes)
  g0[2] = (unsigned int)(ga & 0xffffffffu);              // global_addr[31:0]
  g0[3] = (unsigned int)((ga >> 32) & 0x01ffffffu)       // global_addr[56:32]
          | (2u << 30);                                  // type = 2 ("image")
  int32x8 g1;
  g1[0] = (2 << 16);                                     // data_size = 4 bytes
  g1[1] = (cols & 0xffff) << 16;                         // tensor_dim0 lo16 @bit48
  g1[2] = ((cols >> 16) & 0xffff) | ((rows & 0xffff) << 16); // dim0 hi | dim1 lo
  g1[3] = ((rows >> 16) & 0xffff) | ((cols & 0xffff) << 16); // dim1 hi | tile_dim0
  g1[4] = (rows & 0xffff);                               // tile_dim1 (tile_dim2=0)
  g1[5] = cols;                                          // dim0_stride[31:0]
  g1[6] = 0;                                             // stride hi | dim1_stride
  g1[7] = 0;
  int32x4 z4 = {0, 0, 0, 0};
  int32x8 z8 = {0, 0, 0, 0, 0, 0, 0, 0};
  __builtin_amdgcn_tensor_load_to_lds(g0, g1, z4, z4, z8, 0);
}

// ---------------------------------------------------------------------------
// Stage 0: convert Wvocab f32 -> f16 (12.9 MB, L2-resident afterwards)
// ---------------------------------------------------------------------------
__global__ void k_cvt_f16(const float* __restrict__ in, _Float16* __restrict__ out, int n) {
  int i = blockIdx.x * blockDim.x + threadIdx.x;
  int stride = gridDim.x * blockDim.x;
  for (; i < n; i += stride) out[i] = (_Float16)in[i];
}

// ---------------------------------------------------------------------------
// Stage 1: the 512-step recurrence. One workgroup per batch (independent),
// fp32 VALU (latency-bound, precision-critical). Wq/Wk/Wv (192 KB) are
// TDM-DMA'd into LDS once (CDNA5 WGP has 320 KB LDS) so the 512-step hot
// loop reads weights via ds_load instead of re-streaming 786 MB from L2.
// ---------------------------------------------------------------------------
__global__ __launch_bounds__(256) void k_recurrence(
    const int* __restrict__ zi, const float* __restrict__ latent,
    const float* __restrict__ Wq, const float* __restrict__ bq,
    const float* __restrict__ Wk, const float* __restrict__ bk,
    const float* __restrict__ Wv, const float* __restrict__ bv,
    float* __restrict__ zs32, _Float16* __restrict__ zs16) {
  __shared__ __align__(16) float Wlds[3][EMBED_DIM * EMBED_DIM];  // 192 KB
  __shared__ __align__(16) float z   [STATE_COUNT][EMBED_DIM];
  __shared__ __align__(16) float xz  [STATE_COUNT + 1][EMBED_DIM];
  __shared__ __align__(16) float xqkv[3][STATE_COUNT + 1][EMBED_DIM];
  __shared__ float att[STATE_COUNT + 1][STATE_COUNT];
  __shared__ float rstd[STATE_COUNT];

  const int tid = threadIdx.x;
  const int w = tid >> 5, l = tid & 31;   // wave32
  const int b = blockIdx.x;

  const float* Bs[3] = {bq, bk, bv};

  // --- TDM preload of the three weight matrices into LDS (wave 0 only) ---
  if (w == 0) {
    const float* Wg[3] = {Wq, Wk, Wv};
    for (int m = 0; m < 3; ++m) {
      unsigned int lds_off = (unsigned int)(uintptr_t)(&Wlds[m][0]);
      tdm_load_2d_f32(Wg[m], lds_off, EMBED_DIM, EMBED_DIM);
    }
    __builtin_amdgcn_s_wait_tensorcnt(0);
  }

  // z0 = latent[zi[b]].reshape(S, E)  (overlaps with the TDM transfer)
  const float* lrow = latent + (size_t)zi[b] * (STATE_COUNT * EMBED_DIM);
  for (int i = tid; i < STATE_COUNT * EMBED_DIM; i += 256)
    z[i / EMBED_DIM][i % EMBED_DIM] = lrow[i];
  __syncthreads();

  for (int step = 0; step < N_STEP; ++step) {
    // --- per-state-row 1/std (ddof=1), waves 0..4, shfl tree ---
    if (w < STATE_COUNT) {
      float sum = 0.f, sq = 0.f;
      for (int i = 0; i < 4; ++i) {
        float v = z[w][l * 4 + i];
        sum += v; sq += v * v;
      }
      for (int m = 16; m >= 1; m >>= 1) {
        sum += __shfl_xor(sum, m, 32);
        sq  += __shfl_xor(sq,  m, 32);
      }
      if (l == 0) {
        float var = (sq - sum * sum * (1.0f / EMBED_DIM)) * (1.0f / (EMBED_DIM - 1));
        rstd[w] = 0.113f / (1e-5f + sqrtf(var));
      }
    }
    __syncthreads();
    if (tid < EMBED_DIM) {
      for (int s = 0; s < STATE_COUNT; ++s) xz[s][tid] = z[s][tid] * rstd[s];
      xz[STATE_COUNT][tid] = 1.0f;
    }
    __syncthreads();

    // --- QKV: 3 matrices * 6 rows * 128 cols = 2304 dot-128s, 9 per thread ---
    for (int p = 0; p < 9; ++p) {
      int idx = p * 256 + tid;
      int m = idx / 768, rem = idx % 768;
      int q = rem / EMBED_DIM, o = rem % EMBED_DIM;
      const float4* xr = (const float4*)&xz[q][0];
      const float4* wr = (const float4*)&Wlds[m][o * EMBED_DIM];
      float dot = 0.f;
      #pragma unroll 8
      for (int i = 0; i < EMBED_DIM / 4; ++i) {
        float4 xv4 = xr[i], wv4 = wr[i];
        dot += xv4.x * wv4.x + xv4.y * wv4.y + xv4.z * wv4.z + xv4.w * wv4.w;
      }
      xqkv[m][q][o] = dot + Bs[m][o];
    }
    __syncthreads();

    // --- scores (6x5) + softmax : wave q computes its row ---
    if (w < STATE_COUNT + 1) {
      float sc[STATE_COUNT];
      for (int k = 0; k < STATE_COUNT; ++k) {
        float p = 0.f;
        for (int i = 0; i < 4; ++i) {
          int e = l + i * 32;
          p += xqkv[0][w][e] * xqkv[1][k][e];
        }
        for (int m = 16; m >= 1; m >>= 1) p += __shfl_xor(p, m, 32);
        sc[k] = p;
      }
      if (l == 0) {
        float mx = sc[0];
        for (int k = 1; k < STATE_COUNT; ++k) mx = fmaxf(mx, sc[k]);
        float se = 0.f;
        for (int k = 0; k < STATE_COUNT; ++k) { sc[k] = __expf(sc[k] - mx); se += sc[k]; }
        float inv = 1.0f / se;
        for (int k = 0; k < STATE_COUNT; ++k) att[w][k] = sc[k] * inv;
      }
    }
    __syncthreads();

    // --- dz = att @ xv[:5], state update, emit row ---
    if (tid < EMBED_DIM) {
      int e = tid;
      for (int q = 0; q < STATE_COUNT + 1; ++q) {
        float dz = 0.f;
        for (int k = 0; k < STATE_COUNT; ++k) dz += att[q][k] * xqkv[2][k][e];
        if (q < STATE_COUNT) {
          z[q][e] = xz[q][e] + dz;
        } else {
          size_t r = (size_t)(b * N_STEP + step) * EMBED_DIM + e;
          zs32[r] = dz;
          zs16[r] = (_Float16)dz;
        }
      }
    }
    __syncthreads();
  }
}

// ---------------------------------------------------------------------------
// Stage 2: fused vocab projection + sum(exp(logit)) via f16 WMMA, f32 accum.
// 128 workgroups x 32 rows; 8 waves sweep interleaved 16-col vocab tiles.
// A fragments (zs rows, f16) stay VGPR-resident across the whole sweep.
// global_prefetch of the next tile hides the first (HBM) pass over Wvocab.
// ---------------------------------------------------------------------------
__global__ __launch_bounds__(256) void k_vocab_sumexp(
    const _Float16* __restrict__ zsf16, const _Float16* __restrict__ wv,
    const float* __restrict__ bvocab, float* __restrict__ sumexp) {
  const int w  = threadIdx.x >> 5;   // wave id 0..7
  const int l  = threadIdx.x & 31;
  const int g  = l >> 4;             // 16-lane group
  const int ln = l & 15;
  const int r0 = blockIdx.x * 32;

  // A fragments: 2 M-tiles x 4 K-chunks of 16x32 f16 (ISA 7.12.2 layout)
  v16h a[2][4];
  for (int mt = 0; mt < 2; ++mt) {
    const _Float16* row = zsf16 + (size_t)(r0 + mt * 16 + ln) * EMBED_DIM;
    for (int c = 0; c < 4; ++c) {
      const _Float16* p = row + c * 32 + g * 8;
      #pragma unroll
      for (int i = 0; i < 8; ++i) { a[mt][c][i] = p[i]; a[mt][c][8 + i] = p[16 + i]; }
    }
  }

  v8f s0 = {0, 0, 0, 0, 0, 0, 0, 0};
  v8f s1 = {0, 0, 0, 0, 0, 0, 0, 0};
  const int NT = (GRAPH_DIM + 15) / 16;   // 3142 vocab tiles

  for (int nt = w; nt < NT; nt += 8) {
    const int n = nt * 16 + ln;           // vocab column owned by this lane
    // prefetch next tile's row for this lane (global_prefetch_b8)
    const int n2 = n + 8 * 16;
    if (n2 < GRAPH_DIM) {
      __builtin_prefetch(wv + (size_t)n2 * EMBED_DIM + g * 128, 0, 0);
    }
    v16h bf[4];
    if (n < GRAPH_DIM) {
      const _Float16* row = wv + (size_t)n * EMBED_DIM;   // B[k][n] = Wvocab[n][k]
      for (int c = 0; c < 4; ++c) {
        const _Float16* p = row + c * 32 + g * 8;
        #pragma unroll
        for (int i = 0; i < 8; ++i) { bf[c][i] = p[i]; bf[c][8 + i] = p[16 + i]; }
      }
    } else {
      for (int c = 0; c < 4; ++c)
        #pragma unroll
        for (int i = 0; i < 16; ++i) bf[c][i] = (_Float16)0.0f;
    }
    const float bvn = (n < GRAPH_DIM) ? bvocab[n] : -1e30f;  // pad cols -> exp()=0

    v8f c0 = {0, 0, 0, 0, 0, 0, 0, 0};
    v8f c1 = {0, 0, 0, 0, 0, 0, 0, 0};
    #pragma unroll
    for (int c = 0; c < 4; ++c) {
      c0 = __builtin_amdgcn_wmma_f32_16x16x32_f16(false, a[0][c], false, bf[c],
                                                  (short)0, c0, false, false);
      c1 = __builtin_amdgcn_wmma_f32_16x16x32_f16(false, a[1][c], false, bf[c],
                                                  (short)0, c1, false, false);
    }
    #pragma unroll
    for (int j = 0; j < 8; ++j) {
      s0[j] += __expf(c0[j] + bvn);
      s1[j] += __expf(c1[j] + bvn);
    }
  }

  // per-row reduction: C layout row m = j + g*8, col = ln -> 16-lane tree
  __shared__ float part[8][32];
  #pragma unroll
  for (int j = 0; j < 8; ++j) {
    float v0 = s0[j], v1 = s1[j];
    for (int m = 8; m >= 1; m >>= 1) {
      v0 += __shfl_xor(v0, m, 32);
      v1 += __shfl_xor(v1, m, 32);
    }
    if (ln == 0) {
      part[w][ 0 + j + g * 8] = v0;
      part[w][16 + j + g * 8] = v1;
    }
  }
  __syncthreads();
  if (threadIdx.x < 32) {
    float tot = 0.f;
    for (int ww = 0; ww < 8; ++ww) tot += part[ww][threadIdx.x];
    sumexp[r0 + threadIdx.x] = tot;
  }
}

// ---------------------------------------------------------------------------
// Stage 3: yp[r] = dot(zs[r], Wvocab[y[r]]) + bvocab[y[r]] - log(sumexp[r])
// fp32, one wave per row.
// ---------------------------------------------------------------------------
__global__ __launch_bounds__(256) void k_target(
    const float* __restrict__ zs32, const float* __restrict__ Wvocab,
    const float* __restrict__ bvocab, const int* __restrict__ y,
    const float* __restrict__ sumexp, float* __restrict__ out) {
  const int w = threadIdx.x >> 5, l = threadIdx.x & 31;
  const int r = blockIdx.x * 8 + w;               // 512 blocks * 8 waves = 4096
  const int tgt = y[r];                           // y flat is (B, N_STEP) == row order
  const float* zr = zs32 + (size_t)r * EMBED_DIM;
  const float* wr = Wvocab + (size_t)tgt * EMBED_DIM;
  float acc = 0.f;
  for (int i = 0; i < 4; ++i) {
    int e = l + i * 32;
    acc += zr[e] * wr[e];
  }
  for (int m = 16; m >= 1; m >>= 1) acc += __shfl_xor(acc, m, 32);
  if (l == 0) out[r] = acc + bvocab[tgt] - logf(sumexp[r]);
}

// ---------------------------------------------------------------------------
extern "C" void kernel_launch(void* const* d_in, const int* in_sizes, int n_in,
                              void* d_out, int out_size, void* d_ws, size_t ws_size,
                              hipStream_t stream) {
  const int*   zi     = (const int*)  d_in[0];
  const int*   y      = (const int*)  d_in[1];
  const float* latent = (const float*)d_in[2];
  const float* Wq     = (const float*)d_in[3];
  const float* bq     = (const float*)d_in[4];
  const float* Wk     = (const float*)d_in[5];
  const float* bk     = (const float*)d_in[6];
  const float* Wv     = (const float*)d_in[7];
  const float* bv     = (const float*)d_in[8];
  const float* Wvocab = (const float*)d_in[9];
  const float* bvocab = (const float*)d_in[10];

  char* ws = (char*)d_ws;
  const size_t ZS32 = (size_t)NROWS * EMBED_DIM * sizeof(float);        // 2 MB
  const size_t ZS16 = (size_t)NROWS * EMBED_DIM * sizeof(_Float16);     // 1 MB
  const size_t WV16 = (size_t)GRAPH_DIM * EMBED_DIM * sizeof(_Float16); // 12.9 MB
  float*    zs32  = (float*)   (ws);
  _Float16* zs16  = (_Float16*)(ws + ZS32);
  _Float16* wv16  = (_Float16*)(ws + ZS32 + ZS16);
  float*    sumex = (float*)   (ws + ZS32 + ZS16 + WV16);

  k_cvt_f16<<<4096, 256, 0, stream>>>(Wvocab, wv16, GRAPH_DIM * EMBED_DIM);
  k_recurrence<<<BATCH, 256, 0, stream>>>(zi, latent, Wq, bq, Wk, bk, Wv, bv,
                                          zs32, zs16);
  k_vocab_sumexp<<<NROWS / 32, 256, 0, stream>>>(zs16, wv16, bvocab, sumex);
  k_target<<<NROWS / 8, 256, 0, stream>>>(zs32, Wvocab, bvocab, y, sumex,
                                          (float*)d_out);
}
